// RNNEncoderP_75797582840064
// MI455X (gfx1250) — compile-verified, matrix operand order checked by reading, not run on hardware
//
#include <hip/hip_runtime.h>

// Problem constants (from the reference)
#define D_    128
#define T_    256
#define B_    1024
#define TD_   384   // 3*D
#define NL_   2
#define BN_EPS_ 1e-5f

typedef __attribute__((ext_vector_type(16))) __bf16 bf16x16;
typedef __attribute__((ext_vector_type(8)))  float  f32x8;

__device__ __forceinline__ f32x8 wmma_bf16(bf16x16 a, bf16x16 b, f32x8 c) {
  // v_wmma_f32_16x16x32_bf16: D = A(16x32) * B(32x16) + C(16x16 f32)
  return __builtin_amdgcn_wmma_f32_16x16x32_bf16(false, a, false, b, (short)0, c, false, false);
}

__device__ __forceinline__ float sigmoid_f(float v) {
  return 1.0f / (1.0f + __expf(-v));
}

// A-fragment (16x32 bf16) for lane (half = lane>>4, m = lane&15) built from a
// row-major f32 [16][128] LDS tile.  Per ISA layout: lane holds row m, with
// K = kt*32 + 8*half + {0..7} (vgpr 0-3) and K = kt*32 + 16 + 8*half + {0..7}
// (vgpr 4-7).
__device__ __forceinline__ bf16x16 a_frag_lds(const float* src, int m, int half, int kt) {
  const float4* p = (const float4*)(src + m * D_ + kt * 32 + half * 8);
  float4 c0 = p[0], c1 = p[1], c2 = p[4], c3 = p[5];
  bf16x16 f;
  f[0]=(__bf16)c0.x;  f[1]=(__bf16)c0.y;  f[2]=(__bf16)c0.z;  f[3]=(__bf16)c0.w;
  f[4]=(__bf16)c1.x;  f[5]=(__bf16)c1.y;  f[6]=(__bf16)c1.z;  f[7]=(__bf16)c1.w;
  f[8]=(__bf16)c2.x;  f[9]=(__bf16)c2.y;  f[10]=(__bf16)c2.z; f[11]=(__bf16)c2.w;
  f[12]=(__bf16)c3.x; f[13]=(__bf16)c3.y; f[14]=(__bf16)c3.z; f[15]=(__bf16)c3.w;
  return f;
}

// B-fragment (32x16 bf16) for lane (half, n).  Weights are stored pre-transposed
// as bf16 [384][128] ([n][k], k contiguous), so each lane loads two contiguous
// 16-byte chunks: K = kt*32 + 8*half + {0..7} and +16.
__device__ __forceinline__ bf16x16 b_frag_glb(const __bf16* WT, int ncol, int half, int kt) {
  const uint4* p = (const uint4*)(WT + (size_t)ncol * D_ + kt * 32 + half * 8);
  union { uint4 u[2]; bf16x16 v; } t;
  t.u[0] = p[0];
  t.u[1] = p[2];   // +32 bytes = +16 bf16 elements
  return t.v;
}

// ---------------------------------------------------------------------------
// Kernel 1: transpose + convert GRU weights to bf16 [band][l][n=384][k=128]
// ---------------------------------------------------------------------------
__global__ __launch_bounds__(256)
void prep_weights(const float* __restrict__ Wx_g, const float* __restrict__ Wh_g,
                  const float* __restrict__ Wx_r, const float* __restrict__ Wh_r,
                  __bf16* __restrict__ WxT, __bf16* __restrict__ WhT) {
  int tid = blockIdx.x * blockDim.x + threadIdx.x;
  const int per = 2 * NL_ * TD_ * D_;   // 196608
  if (tid >= per) return;
  int k = tid & (D_ - 1);
  int rest = tid >> 7;                  // (band*2 + l)*384 + n
  int n = rest % TD_;
  rest /= TD_;
  int l = rest & 1;
  int band = rest >> 1;
  const float* Wx = band ? Wx_r : Wx_g;
  const float* Wh = band ? Wh_r : Wh_g;
  WxT[tid] = (__bf16)Wx[(size_t)(l * D_ + k) * TD_ + n];
  WhT[tid] = (__bf16)Wh[(size_t)(l * D_ + k) * TD_ + n];
}

// ---------------------------------------------------------------------------
// Kernel 2: full GRU scan + online masked-softmax attention pooling.
// grid = 2 bands * 64 tiles, block = 1 wave (32 threads) handling 16 batch rows.
// The N-tile loop is a runtime loop (unroll 1) so the scheduler can only hoist
// one tile's worth of streamed weight loads -> no scratch spills.
// ---------------------------------------------------------------------------
__global__ __launch_bounds__(32)
void scan_kernel(const float* __restrict__ xg,  const float* __restrict__ dtg,
                 const unsigned char* __restrict__ mkg,
                 const float* __restrict__ Wpg, const float* __restrict__ bxg,
                 const float* __restrict__ bhg, const float* __restrict__ swg,
                 const float* __restrict__ sbg,
                 const float* __restrict__ xr,  const float* __restrict__ dtr,
                 const unsigned char* __restrict__ mkr,
                 const float* __restrict__ Wpr, const float* __restrict__ bxr,
                 const float* __restrict__ bhr, const float* __restrict__ swr,
                 const float* __restrict__ sbr,
                 const __bf16* __restrict__ WxT, const __bf16* __restrict__ WhT,
                 float* __restrict__ zbuf) {
  __shared__ float lds_cur[16 * D_];   // layer input (e_t, then h0_new)
  __shared__ float lds_h0 [16 * D_];
  __shared__ float lds_h1 [16 * D_];
  __shared__ float lds_att[16 * D_];   // online attention accumulator
  __shared__ float lds_br [NL_ * D_];  // bx_r+bh_r
  __shared__ float lds_bz [NL_ * D_];  // bx_z+bh_z
  __shared__ float lds_bxn[NL_ * D_];
  __shared__ float lds_bhn[NL_ * D_];
  __shared__ float lds_sw [D_];        // attention weight vector
  __shared__ float lds_xv[16], lds_dv[16], lds_mk[16];
  __shared__ float lds_part[16][16];
  __shared__ float lds_m[16], lds_ssum[16], lds_scale[16], lds_w[16];

  const int lane = threadIdx.x;
  const int band = blockIdx.x >> 6;
  const int tile = blockIdx.x & 63;
  const int b0   = tile * 16;
  const int half = lane >> 4;
  const int nn   = lane & 15;

  const float* x  = band ? xr  : xg;
  const float* dt = band ? dtr : dtg;
  const unsigned char* mk = band ? mkr : mkg;
  const float* Wp = band ? Wpr : Wpg;
  const float* bx = band ? bxr : bxg;
  const float* bh = band ? bhr : bhg;
  const float* sw = band ? swr : swg;
  const float  sb = band ? sbr[0] : sbg[0];

  const __bf16* WxTb = WxT + (size_t)band * NL_ * TD_ * D_;
  const __bf16* WhTb = WhT + (size_t)band * NL_ * TD_ * D_;

  // warm the caches with the weights this wave will stream 256 times
  for (int off = lane * 64; off < NL_ * TD_ * D_; off += 32 * 64) {
    __builtin_prefetch(WxTb + off, 0, 3);
    __builtin_prefetch(WhTb + off, 0, 3);
  }

  for (int i = lane; i < 16 * D_; i += 32) {
    lds_h0[i] = 0.f; lds_h1[i] = 0.f; lds_att[i] = 0.f;
  }
  for (int i = lane; i < NL_ * D_; i += 32) {
    int l = i >> 7, c = i & (D_ - 1);
    lds_br[i]  = bx[l * TD_ + c]            + bh[l * TD_ + c];
    lds_bz[i]  = bx[l * TD_ + D_ + c]       + bh[l * TD_ + D_ + c];
    lds_bxn[i] = bx[l * TD_ + 2 * D_ + c];
    lds_bhn[i] = bh[l * TD_ + 2 * D_ + c];
  }
  for (int i = lane; i < D_; i += 32) lds_sw[i] = sw[i];
  if (lane < 16) { lds_m[lane] = -3.0e38f; lds_ssum[lane] = 0.f; }

  // small per-lane constants
  float wp0[4], wp1[4];
  #pragma unroll
  for (int c = 0; c < 4; ++c) { wp0[c] = Wp[lane + 32 * c]; wp1[c] = Wp[D_ + lane + 32 * c]; }

  __syncthreads();

  #pragma unroll 1
  for (int t = 0; t < T_; ++t) {
    if (lane < 16) {
      int bi = b0 + lane;
      lds_xv[lane] = x[(size_t)bi * T_ + t];
      lds_dv[lane] = dt[(size_t)bi * T_ + t];
      lds_mk[lane] = mk[(size_t)bi * T_ + t] ? 1.f : 0.f;
    }
    __syncthreads();

    // per-row mask for this step (rows e + 8*half)
    float mrow[8];
    #pragma unroll
    for (int e = 0; e < 8; ++e) mrow[e] = lds_mk[e + half * 8];

    // e_t = x*Wp[0,:] + dtime*Wp[1,:]  -> lds_cur (f32 [16][128])
    #pragma unroll
    for (int ro = 0; ro < 16; ++ro) {
      float xv = lds_xv[ro], dv = lds_dv[ro];
      #pragma unroll
      for (int c = 0; c < 4; ++c)
        lds_cur[ro * D_ + lane + 32 * c] = xv * wp0[c] + dv * wp1[c];
    }
    __syncthreads();

    float part[8] = {0.f, 0.f, 0.f, 0.f, 0.f, 0.f, 0.f, 0.f};

    #pragma unroll
    for (int l = 0; l < NL_; ++l) {
      float* hsrc = (l == 0) ? lds_h0 : lds_h1;
      const __bf16* WxTl = WxTb + (size_t)l * TD_ * D_;
      const __bf16* WhTl = WhTb + (size_t)l * TD_ * D_;

      // A fragments: 4 K-tiles for the layer input and the hidden state.
      bf16x16 a_x[4], a_h[4];
      #pragma unroll
      for (int kt = 0; kt < 4; ++kt) {
        a_x[kt] = a_frag_lds(lds_cur, nn, half, kt);
        a_h[kt] = a_frag_lds(hsrc,    nn, half, kt);
      }

      // Runtime loop over the 8 output-column tiles: bounds scheduler hoisting
      // to one tile's 24 weight loads -> no scratch spills.
      #pragma unroll 1
      for (int j = 0; j < 8; ++j) {
        const int c = j * 16 + nn;
        // biases / attention weight from LDS (scalar DS loads)
        const float brv  = lds_br [l * D_ + c];
        const float bzv  = lds_bz [l * D_ + c];
        const float bxnv = lds_bxn[l * D_ + c];
        const float bhnv = lds_bhn[l * D_ + c];
        const float swv  = lds_sw [c];

        f32x8 ar  = (f32x8){0.f,0.f,0.f,0.f,0.f,0.f,0.f,0.f};
        f32x8 az  = ar, axn = ar, ahn = ar;
        #pragma unroll
        for (int kt = 0; kt < 4; ++kt) {
          ar  = wmma_bf16(a_x[kt], b_frag_glb(WxTl, c,            half, kt), ar);
          ar  = wmma_bf16(a_h[kt], b_frag_glb(WhTl, c,            half, kt), ar);
          az  = wmma_bf16(a_x[kt], b_frag_glb(WxTl, D_ + c,       half, kt), az);
          az  = wmma_bf16(a_h[kt], b_frag_glb(WhTl, D_ + c,       half, kt), az);
          axn = wmma_bf16(a_x[kt], b_frag_glb(WxTl, 2 * D_ + c,   half, kt), axn);
          ahn = wmma_bf16(a_h[kt], b_frag_glb(WhTl, 2 * D_ + c,   half, kt), ahn);
        }
        // GRU pointwise update (C/D layout: row = e + 8*half, col = j*16 + nn)
        #pragma unroll
        for (int e = 0; e < 8; ++e) {
          int idx = (e + half * 8) * D_ + c;
          float rg = sigmoid_f(ar[e] + brv);
          float zg = sigmoid_f(az[e] + bzv);
          float ng = tanhf(axn[e] + bxnv + rg * (ahn[e] + bhnv));
          float hp = hsrc[idx];
          float m  = mrow[e];
          float hn = m * ((1.f - zg) * ng + zg * hp) + (1.f - m) * hp;
          hsrc[idx] = hn;
          if (l == 0) lds_cur[idx] = hn;     // layer-1 input
          else        part[e] += hn * swv;   // attention score partial
        }
      }
      __syncthreads();
    }

    // deterministic cross-lane score reduction
    #pragma unroll
    for (int e = 0; e < 8; ++e) lds_part[e + half * 8][nn] = part[e];
    __syncthreads();
    if (lane < 16) {
      float s = sb;
      #pragma unroll
      for (int i = 0; i < 16; ++i) s += lds_part[lane][i];
      if (lds_mk[lane] > 0.5f) {            // masked step: weight 0, no rescale
        float mo = lds_m[lane];
        float mn = fmaxf(mo, s);
        float sc = __expf(mo - mn);
        float w  = __expf(s - mn);
        lds_ssum[lane]  = lds_ssum[lane] * sc + w;
        lds_m[lane]     = mn;
        lds_scale[lane] = sc;
        lds_w[lane]     = w;
      } else { lds_scale[lane] = 1.f; lds_w[lane] = 0.f; }
    }
    __syncthreads();

    // online attention accumulator update: acc = acc*scale + w * y_t  (in LDS)
    float sc8[8], w8[8];
    #pragma unroll
    for (int e = 0; e < 8; ++e) {
      sc8[e] = lds_scale[e + half * 8];
      w8[e]  = lds_w[e + half * 8];
    }
    #pragma unroll
    for (int j = 0; j < 8; ++j)
      #pragma unroll
      for (int e = 0; e < 8; ++e) {
        int idx = (e + half * 8) * D_ + j * 16 + nn;
        lds_att[idx] = lds_att[idx] * sc8[e] + w8[e] * lds_h1[idx];
      }
    __syncthreads();
  }

  // finalize: z[b,d] = acc / ssum
  float inv8[8];
  #pragma unroll
  for (int e = 0; e < 8; ++e) inv8[e] = 1.0f / lds_ssum[e + half * 8];
  #pragma unroll
  for (int j = 0; j < 8; ++j)
    #pragma unroll
    for (int e = 0; e < 8; ++e) {
      int row = e + half * 8;
      int col = j * 16 + nn;
      zbuf[(size_t)band * B_ * D_ + (size_t)(b0 + row) * D_ + col] =
          lds_att[row * D_ + col] * inv8[e];
    }
}

// ---------------------------------------------------------------------------
// Kernel 3: [1024,256] @ [256,256] mb projection via WMMA
// ---------------------------------------------------------------------------
__global__ __launch_bounds__(32)
void mb_gemm(const float* __restrict__ zbuf, const float* __restrict__ mbW,
             float* __restrict__ mbout) {
  const int lane = threadIdx.x;
  const int half = lane >> 4;
  const int nn   = lane & 15;
  const int b0 = blockIdx.x * 16;
  const int c0 = blockIdx.y * 16;
  const float* zg = zbuf;
  const float* zr = zbuf + (size_t)B_ * D_;

  f32x8 acc = (f32x8){0.f,0.f,0.f,0.f,0.f,0.f,0.f,0.f};
  #pragma unroll
  for (int kt = 0; kt < 8; ++kt) {
    const float* src = (kt < 4) ? zg : zr;
    int kbase = (kt & 3) * 32 + half * 8;
    bf16x16 a;
    const float* ap = src + (size_t)(b0 + nn) * D_ + kbase;
    #pragma unroll
    for (int i = 0; i < 8; ++i) { a[i] = (__bf16)ap[i]; a[i + 8] = (__bf16)ap[i + 16]; }
    bf16x16 b;
    int col = c0 + nn;
    #pragma unroll
    for (int i = 0; i < 8; ++i) {
      int k1 = kt * 32 + half * 8 + i;
      b[i]     = (__bf16)mbW[(size_t)k1 * 256 + col];
      b[i + 8] = (__bf16)mbW[(size_t)(k1 + 16) * 256 + col];
    }
    acc = wmma_bf16(a, b, acc);
  }
  #pragma unroll
  for (int e = 0; e < 8; ++e)
    mbout[(size_t)(b0 + e + half * 8) * 256 + (c0 + nn)] = acc[e];
}

// ---------------------------------------------------------------------------
// Kernel 4: BatchNorm1d over batch (deterministic tree reduce per column)
// ---------------------------------------------------------------------------
__global__ __launch_bounds__(256)
void bn_kernel(const float* __restrict__ mbout, const float* __restrict__ gamma,
               const float* __restrict__ beta, float* __restrict__ out) {
  __shared__ float s1[256], s2[256];
  const int col = blockIdx.x;
  const int tid = threadIdx.x;
  float v[4], sum = 0.f, sq = 0.f;
  #pragma unroll
  for (int i = 0; i < 4; ++i) {
    v[i] = mbout[(size_t)(tid + 256 * i) * 256 + col];
    sum += v[i]; sq += v[i] * v[i];
  }
  s1[tid] = sum; s2[tid] = sq;
  __syncthreads();
  for (int off = 128; off > 0; off >>= 1) {
    if (tid < off) { s1[tid] += s1[tid + off]; s2[tid] += s2[tid + off]; }
    __syncthreads();
  }
  float mu  = s1[0] * (1.f / 1024.f);
  float var = s2[0] * (1.f / 1024.f) - mu * mu;
  float inv = rsqrtf(var + BN_EPS_);
  float g = gamma[col], bb = beta[col];
  #pragma unroll
  for (int i = 0; i < 4; ++i)
    out[(size_t)(tid + 256 * i) * 256 + col] = (v[i] - mu) * inv * g + bb;
}

// ---------------------------------------------------------------------------
// host launcher
// ---------------------------------------------------------------------------
extern "C" void kernel_launch(void* const* d_in, const int* in_sizes, int n_in,
                              void* d_out, int out_size, void* d_ws, size_t ws_size,
                              hipStream_t stream) {
  (void)in_sizes; (void)n_in; (void)out_size; (void)ws_size;
  const float* x_g  = (const float*)d_in[0];
  const float* dt_g = (const float*)d_in[1];
  const unsigned char* mk_g = (const unsigned char*)d_in[2];
  const float* Wp_g = (const float*)d_in[3];
  const float* Wx_g = (const float*)d_in[4];
  const float* Wh_g = (const float*)d_in[5];
  const float* bx_g = (const float*)d_in[6];
  const float* bh_g = (const float*)d_in[7];
  const float* sw_g = (const float*)d_in[8];
  const float* sb_g = (const float*)d_in[9];
  const float* x_r  = (const float*)d_in[10];
  const float* dt_r = (const float*)d_in[11];
  const unsigned char* mk_r = (const unsigned char*)d_in[12];
  const float* Wp_r = (const float*)d_in[13];
  const float* Wx_r = (const float*)d_in[14];
  const float* Wh_r = (const float*)d_in[15];
  const float* bx_r = (const float*)d_in[16];
  const float* bh_r = (const float*)d_in[17];
  const float* sw_r = (const float*)d_in[18];
  const float* sb_r = (const float*)d_in[19];
  const float* mbW  = (const float*)d_in[20];
  const float* gam  = (const float*)d_in[21];
  const float* bet  = (const float*)d_in[22];

  // workspace layout (needs ~2.75 MB):
  //   [0,        393216)  WxT bf16 [2][2][384][128]
  //   [393216,   786432)  WhT bf16
  //   [786432,  1835008)  zbuf f32 [2][1024][128]
  //   [1835008, 2883584)  mbout f32 [1024][256]
  char* ws = (char*)d_ws;
  __bf16* WxT  = (__bf16*)ws;
  __bf16* WhT  = (__bf16*)(ws + 393216);
  float*  zbuf = (float*)(ws + 786432);
  float*  mbout= (float*)(ws + 1835008);

  prep_weights<<<dim3(768), dim3(256), 0, stream>>>(Wx_g, Wh_g, Wx_r, Wh_r, WxT, WhT);
  scan_kernel<<<dim3(128), dim3(32), 0, stream>>>(
      x_g, dt_g, mk_g, Wp_g, bx_g, bh_g, sw_g, sb_g,
      x_r, dt_r, mk_r, Wp_r, bx_r, bh_r, sw_r, sb_r,
      WxT, WhT, zbuf);
  mb_gemm<<<dim3(64, 16), dim3(32), 0, stream>>>(zbuf, mbW, mbout);
  bn_kernel<<<dim3(256), dim3(256), 0, stream>>>(mbout, gam, bet, (float*)d_out);
}
